// GPTModel_61168924229756
// MI455X (gfx1250) — matrix operand order repaired and checked by reading, not compiled, not run on hardware
//
#include <hip/hip_runtime.h>
#include <math.h>

// CDNA5 WMMA vector types (probe-confirmed)
typedef __attribute__((ext_vector_type(16))) __bf16 v16bf;
typedef __attribute__((ext_vector_type(4)))  __bf16 v4bf;
typedef __attribute__((ext_vector_type(8)))  float  v8f;

#define EPS 1e-5f
#define LROW 72   // 64 + 8 bf16 pad (row stride 144B, 16B-aligned chunks)

static __device__ __forceinline__ float gelu_exact(float x) {
    return 0.5f * x * (1.0f + erff(x * 0.70710678118654752f));
}

// CDNA5 async global->LDS copy (16B per lane), tracked by ASYNCcnt.
static __device__ __forceinline__ void async_ld_b128(unsigned lds_addr,
                                                     unsigned long long gaddr) {
    asm volatile("global_load_async_to_lds_b128 %0, %1, off"
                 :: "v"(lds_addr), "v"(gaddr) : "memory");
}
static __device__ __forceinline__ void wait_async0() {
    asm volatile("s_wait_asynccnt 0x0" ::: "memory");
}

// One K=64 stage of compute from a fixed (compile-time) LDS buffer.
// Wave computes 4 adjacent 16x16 N-tiles sharing one A fragment per K-slice.
static __device__ __forceinline__ void compute_stage(
    const __bf16 (* __restrict__ la)[LROW],
    const __bf16 (* __restrict__ lb)[LROW],
    int mrow, int ncol, int rowA, int khalf, int colB, int ksegB,
    v8f acc[4])
{
    #pragma unroll
    for (int kk = 0; kk < 64; kk += 32) {
        // A fragment (16x32, ISA 7.12.2 16-bit A layout)
        v16bf af;
        const __bf16* ar = &la[mrow + rowA][kk];
        #pragma unroll
        for (int e = 0; e < 8; ++e) af[e]     = ar[khalf + e];
        #pragma unroll
        for (int e = 0; e < 8; ++e) af[8 + e] = ar[16 + khalf + e];

        #pragma unroll
        for (int j = 0; j < 4; ++j) {
            // B fragment (32x16): lane = column, elements = K in segment
            v16bf bf;
            const __bf16* bp = &lb[ncol + j * 16 + colB][kk + ksegB];
            #pragma unroll
            for (int e = 0; e < 8; ++e) { bf[e] = bp[e]; bf[8 + e] = bp[16 + e]; }
            acc[j] = __builtin_amdgcn_wmma_f32_16x16x32_bf16(
                false, af, false, bf, (short)0, acc[j], false, false);
        }
    }
}

// ----------------------------------------------------------------------------
// f32 -> bf16 conversion (float4 in, packed 4x bf16 out). n % 1024 == 0.
// ----------------------------------------------------------------------------
__global__ __launch_bounds__(256) void cvt_bf16_kernel(
    const float* __restrict__ in, __bf16* __restrict__ out)
{
    int i = blockIdx.x * 256 + threadIdx.x;       // float4 slot
    float4 f = ((const float4*)in)[i];
    v4bf p; p[0] = (__bf16)f.x; p[1] = (__bf16)f.y;
            p[2] = (__bf16)f.z; p[3] = (__bf16)f.w;
    ((v4bf*)out)[i] = p;
}

// ----------------------------------------------------------------------------
// Embedding: h[b,s,:] = tok_emb[x[b,s],:] + pos_emb[s,:]
// ----------------------------------------------------------------------------
__global__ __launch_bounds__(256) void embed_kernel(
    const int* __restrict__ x, const float* __restrict__ tok,
    const float* __restrict__ pos, float* __restrict__ h, int S, int D)
{
    int idx = blockIdx.x * 256 + threadIdx.x;           // over BS*D
    int row = idx / D;                                   // b*S + s
    int d   = idx - row * D;
    int s   = row % S;
    int t   = x[row];
    h[idx] = tok[(size_t)t * D + d] + pos[(size_t)s * D + d];
}

// ----------------------------------------------------------------------------
// LayerNorm: one block per row of length D
// ----------------------------------------------------------------------------
__global__ __launch_bounds__(256) void ln_kernel(
    const float* __restrict__ in, const float* __restrict__ g,
    const float* __restrict__ b, float* __restrict__ out, int D)
{
    __shared__ float r1[256], r2[256];
    int row = blockIdx.x;
    const float* xr = in + (size_t)row * D;
    float s = 0.f, s2 = 0.f;
    for (int d = threadIdx.x; d < D; d += 256) {
        float v = xr[d]; s += v; s2 += v * v;
    }
    r1[threadIdx.x] = s; r2[threadIdx.x] = s2;
    __syncthreads();
    for (int off = 128; off > 0; off >>= 1) {
        if (threadIdx.x < off) {
            r1[threadIdx.x] += r1[threadIdx.x + off];
            r2[threadIdx.x] += r2[threadIdx.x + off];
        }
        __syncthreads();
    }
    float mu  = r1[0] / (float)D;
    float var = r2[0] / (float)D - mu * mu;
    float inv = rsqrtf(var + EPS);
    float* orow = out + (size_t)row * D;
    for (int d = threadIdx.x; d < D; d += 256)
        orow[d] = (xr[d] - mu) * inv * g[d] + b[d];
}

// ----------------------------------------------------------------------------
// GEMM: C[M,N] = A[M,K] @ W[N,K]^T (+ bias[N]) with optional epilogue.
//   A, W are bf16 (pre-converted); C/bias/res are f32.
//   epi = 0: none, 1: exact GELU, 2: += res[M,N]
// Requirements: M % 64 == 0, N % 128 == 0, K % 128 == 0 (hold for this model).
// 256 threads = 8 wave32 (4 M-groups x 2 N-groups); block tile 64x128x(K=64).
// Tiles staged with global_load_async_to_lds_b128, double-buffered LDS; stage
// loop unrolled by 2 so buffer indices are compile-time constants:
//   wait_async -> barrier -> issue next tile (other buf) -> 8 WMMAs (this buf)
// ----------------------------------------------------------------------------
__global__ __launch_bounds__(256) void gemm_bf16_wmma(
    const __bf16* __restrict__ A, const __bf16* __restrict__ W,
    const float* __restrict__ bias, const float* __restrict__ res,
    float* __restrict__ C, int M, int N, int K, int epi)
{
    __shared__ __bf16 lA[2][64][LROW];    // 18.4 KB
    __shared__ __bf16 lB[2][128][LROW];   // 36.9 KB

    const int tid   = threadIdx.x;
    const int lane  = tid & 31;
    const int wave  = tid >> 5;           // 0..7
    const int mrow  = (wave & 3) * 16;    // wave's M sub-tile
    const int ncol  = (wave >> 2) * 64;   // wave's N sub-block (64 wide)
    const int mBase = blockIdx.y * 64;
    const int nBase = blockIdx.x * 128;

    v8f acc[4] = {};

    // async staging: A = 512 16B chunks (2/thread), B = 1024 (4/thread)
    unsigned laddrA[2][2], laddrB[2][4];
    unsigned long long gA[2], gB[4];
    #pragma unroll
    for (int c = 0; c < 2; ++c) {
        int idx = tid + c * 256;                   // 0..511
        int r = idx >> 3, c16 = idx & 7;
        laddrA[0][c] = (unsigned)(size_t)&lA[0][r][c16 * 8];
        laddrA[1][c] = (unsigned)(size_t)&lA[1][r][c16 * 8];
        gA[c] = (unsigned long long)(size_t)(A + (size_t)(mBase + r) * K + c16 * 8);
    }
    #pragma unroll
    for (int c = 0; c < 4; ++c) {
        int idx = tid + c * 256;                   // 0..1023
        int r = idx >> 3, c16 = idx & 7;
        laddrB[0][c] = (unsigned)(size_t)&lB[0][r][c16 * 8];
        laddrB[1][c] = (unsigned)(size_t)&lB[1][r][c16 * 8];
        gB[c] = (unsigned long long)(size_t)(W + (size_t)(nBase + r) * K + c16 * 8);
    }

#define ISSUE_TILE(buf)                                        \
    do {                                                       \
        _Pragma("unroll")                                      \
        for (int c = 0; c < 2; ++c) async_ld_b128(laddrA[buf][c], gA[c]); \
        _Pragma("unroll")                                      \
        for (int c = 0; c < 4; ++c) async_ld_b128(laddrB[buf][c], gB[c]); \
    } while (0)
#define ADVANCE_K()                                            \
    do {                                                       \
        _Pragma("unroll")                                      \
        for (int c = 0; c < 2; ++c) gA[c] += 128;              \
        _Pragma("unroll")                                      \
        for (int c = 0; c < 4; ++c) gB[c] += 128;              \
    } while (0)

    ISSUE_TILE(0);                         // prologue: stage 0 -> buffer 0

    const int rowA  = lane & 15;
    const int khalf = (lane >> 4) * 8;     // A: 8-wide K half within each 16
    const int colB  = lane & 15;
    const int ksegB = (lane >> 4) * 8;     // B: 8-wide K half within each 16

    const int nstage = K >> 6;             // K/64, even (K % 128 == 0)
    for (int s = 0; s < nstage; s += 2) {
        wait_async0();
        __syncthreads();
        if (s + 1 < nstage) { ADVANCE_K(); ISSUE_TILE(1); }
        compute_stage(lA[0], lB[0], mrow, ncol, rowA, khalf, colB, ksegB, acc);

        wait_async0();
        __syncthreads();
        if (s + 2 < nstage) { ADVANCE_K(); ISSUE_TILE(0); }
        compute_stage(lA[1], lB[1], mrow, ncol, rowA, khalf, colB, ksegB, acc);
    }
#undef ISSUE_TILE
#undef ADVANCE_K

    // C layout: VGPR v -> row v + 8*(lane>=16); lane&15 -> column
    const int r0 = mBase + mrow + (lane >> 4) * 8;
    #pragma unroll
    for (int j = 0; j < 4; ++j) {
        int cc = nBase + ncol + j * 16 + (lane & 15);
        float bv = bias ? bias[cc] : 0.0f;
        #pragma unroll
        for (int v = 0; v < 8; ++v) {
            int r = r0 + v;
            float val = acc[j][v] + bv;
            if (epi == 1)      val = gelu_exact(val);
            else if (epi == 2) val += res[(size_t)r * N + cc];
            C[(size_t)r * N + cc] = val;
        }
    }
}

// ----------------------------------------------------------------------------
// Attention: one block per (b, h, q). qkv rows are [q(D) | k(D) | v(D)].
// Causal softmax over keys 0..q, then weighted sum of V (HD = 64).
// ----------------------------------------------------------------------------
__global__ __launch_bounds__(256) void attn_kernel(
    const float* __restrict__ qkv, float* __restrict__ out,
    int S, int D, int H)
{
    __shared__ float sc[1024];
    __shared__ float qs[64];
    __shared__ float red[256];

    const int tid = threadIdx.x;
    const int q   = blockIdx.x % S;
    const int hh  = (blockIdx.x / S) % H;
    const int b   = blockIdx.x / (S * H);
    const int D3  = 3 * D;
    const float scale = 0.125f;  // 1/sqrt(64)

    const float* qrow = qkv + (size_t)(b * S + q) * D3 + hh * 64;
    if (tid < 64) qs[tid] = qrow[tid];
    __syncthreads();

    float lmax = -3.4e38f;
    for (int k = tid; k <= q; k += 256) {
        const float* krow = qkv + (size_t)(b * S + k) * D3 + D + hh * 64;
        float s = 0.f;
        #pragma unroll
        for (int d = 0; d < 64; ++d) s += qs[d] * krow[d];
        s *= scale;
        sc[k] = s;
        lmax = fmaxf(lmax, s);
    }
    red[tid] = lmax; __syncthreads();
    for (int off = 128; off > 0; off >>= 1) {
        if (tid < off) red[tid] = fmaxf(red[tid], red[tid + off]);
        __syncthreads();
    }
    float mx = red[0]; __syncthreads();

    float lsum = 0.f;
    for (int k = tid; k <= q; k += 256) {
        float p = __expf(sc[k] - mx);
        sc[k] = p;
        lsum += p;
    }
    red[tid] = lsum; __syncthreads();
    for (int off = 128; off > 0; off >>= 1) {
        if (tid < off) red[tid] += red[tid + off];
        __syncthreads();
    }
    float inv = 1.0f / red[0];

    if (tid < 64) {
        const float* vcol = qkv + (size_t)(b * S) * D3 + 2 * D + hh * 64 + tid;
        float o = 0.f;
        for (int k = 0; k <= q; ++k) o += sc[k] * vcol[(size_t)k * D3];
        out[(size_t)(b * S + q) * D + hh * 64 + tid] = o * inv;
    }
}

// ----------------------------------------------------------------------------
// Host dispatch
// ----------------------------------------------------------------------------
extern "C" void kernel_launch(void* const* d_in, const int* in_sizes, int n_in,
                              void* d_out, int out_size, void* d_ws, size_t ws_size,
                              hipStream_t stream)
{
    const int B = 2, S = 1024, D = 1024, H = 16, L = 6, FF = 4096, V = 32000;
    const int BS = B * S;

    const int*   x      = (const int*)  d_in[0];
    const float* tok    = (const float*)d_in[1];
    const float* pos    = (const float*)d_in[2];
    const float* ln1_g  = (const float*)d_in[3];
    const float* ln1_b  = (const float*)d_in[4];
    const float* qkv_w  = (const float*)d_in[5];
    const float* qkv_b  = (const float*)d_in[6];
    const float* proj_w = (const float*)d_in[7];
    const float* proj_b = (const float*)d_in[8];
    const float* ln2_g  = (const float*)d_in[9];
    const float* ln2_b  = (const float*)d_in[10];
    const float* fc1_w  = (const float*)d_in[11];
    const float* fc1_b  = (const float*)d_in[12];
    const float* fc2_w  = (const float*)d_in[13];
    const float* fc2_b  = (const float*)d_in[14];
    const float* lnf_g  = (const float*)d_in[15];
    const float* lnf_b  = (const float*)d_in[16];
    const float* head_w = (const float*)d_in[17];

    float* ws   = (float*)d_ws;
    float* h    = ws;                               // BS*D f32
    float* xn   = h   + (size_t)BS * D;             // BS*D f32
    float* qkv  = xn  + (size_t)BS * D;             // BS*3D f32
    float* att  = qkv + (size_t)BS * 3 * D;         // BS*D f32
    float* ff   = att + (size_t)BS * D;             // BS*FF f32
    __bf16* abf = (__bf16*)(ff + (size_t)BS * FF);  // BS*FF bf16 (max A size)
    __bf16* wbf = abf + (size_t)BS * FF;            // V*D bf16 (max W size)

    embed_kernel<<<dim3(BS * D / 256), dim3(256), 0, stream>>>(x, tok, pos, h, S, D);

    for (int l = 0; l < L; ++l) {
        // ---- attention sub-block ----
        ln_kernel<<<dim3(BS), dim3(256), 0, stream>>>(
            h, ln1_g + (size_t)l * D, ln1_b + (size_t)l * D, xn, D);
        cvt_bf16_kernel<<<dim3(BS * D / 1024), dim3(256), 0, stream>>>(xn, abf);
        cvt_bf16_kernel<<<dim3(3 * D * D / 1024), dim3(256), 0, stream>>>(
            qkv_w + (size_t)l * 3 * D * D, wbf);
        gemm_bf16_wmma<<<dim3(3 * D / 128, BS / 64), dim3(256), 0, stream>>>(
            abf, wbf, qkv_b + (size_t)l * 3 * D, nullptr, qkv,
            BS, 3 * D, D, /*epi=*/0);

        attn_kernel<<<dim3(B * H * S), dim3(256), 0, stream>>>(qkv, att, S, D, H);

        cvt_bf16_kernel<<<dim3(BS * D / 1024), dim3(256), 0, stream>>>(att, abf);
        cvt_bf16_kernel<<<dim3(D * D / 1024), dim3(256), 0, stream>>>(
            proj_w + (size_t)l * D * D, wbf);
        gemm_bf16_wmma<<<dim3(D / 128, BS / 64), dim3(256), 0, stream>>>(
            abf, wbf, proj_b + (size_t)l * D, h, h, BS, D, D, /*epi=*/2);

        // ---- MLP sub-block ----
        ln_kernel<<<dim3(BS), dim3(256), 0, stream>>>(
            h, ln2_g + (size_t)l * D, ln2_b + (size_t)l * D, xn, D);
        cvt_bf16_kernel<<<dim3(BS * D / 1024), dim3(256), 0, stream>>>(xn, abf);
        cvt_bf16_kernel<<<dim3(FF * D / 1024), dim3(256), 0, stream>>>(
            fc1_w + (size_t)l * FF * D, wbf);
        gemm_bf16_wmma<<<dim3(FF / 128, BS / 64), dim3(256), 0, stream>>>(
            abf, wbf, fc1_b + (size_t)l * FF, nullptr, ff, BS, FF, D, /*epi=*/1);

        cvt_bf16_kernel<<<dim3(BS * FF / 1024), dim3(256), 0, stream>>>(ff, abf);
        cvt_bf16_kernel<<<dim3(D * FF / 1024), dim3(256), 0, stream>>>(
            fc2_w + (size_t)l * D * FF, wbf);
        gemm_bf16_wmma<<<dim3(D / 128, BS / 64), dim3(256), 0, stream>>>(
            abf, wbf, fc2_b + (size_t)l * D, h, h, BS, D, FF, /*epi=*/2);
    }

    // ---- final LN + LM head ----
    ln_kernel<<<dim3(BS), dim3(256), 0, stream>>>(h, lnf_g, lnf_b, xn, D);
    cvt_bf16_kernel<<<dim3(BS * D / 1024), dim3(256), 0, stream>>>(xn, abf);
    cvt_bf16_kernel<<<dim3(V * D / 1024), dim3(256), 0, stream>>>(head_w, wbf);
    gemm_bf16_wmma<<<dim3(V / 128, BS / 64), dim3(256), 0, stream>>>(
        abf, wbf, nullptr, nullptr, (float*)d_out, BS, V, D, /*epi=*/0);
}
